// BahdanauAttention_76553497084057
// MI455X (gfx1250) — matrix operand order repaired and checked by reading, not compiled, not run on hardware
//
#include <hip/hip_runtime.h>
#include <hip/hip_bf16.h>
#include <math.h>

typedef __attribute__((ext_vector_type(16))) __bf16 v16bf;
typedef __attribute__((ext_vector_type(8)))  float  v8f;

#define NB   32      // batch
#define NS   4096    // sequence
#define ND   512     // ENC = DEC = ATT
#define MT   64      // s-rows per workgroup in score kernel (4 WMMA row tiles)

// ---------- helpers ----------
__device__ __forceinline__ v16bf make_frag(const __bf16* p0, const __bf16* p1) {
    union { uint4 u[2]; v16bf v; } x;
    x.u[0] = *(const uint4*)p0;   // 8 contiguous bf16
    x.u[1] = *(const uint4*)p1;   // next 8 contiguous bf16
    return x.v;
}

struct bf16x4 { __bf16 x, y, z, w; };

// branch-free tanh: native v_tanh if available, else exp/rcp formulation.
// tanh(x) = 1 - 2/(exp(2x)+1); exp->inf => 1, exp->0 => -1, so saturation is correct.
__device__ __forceinline__ float fast_tanh(float x) {
#if __has_builtin(__builtin_amdgcn_tanhf)
    return __builtin_amdgcn_tanhf(x);
#elif __has_builtin(__builtin_amdgcn_tanh_f32)
    return __builtin_amdgcn_tanh_f32(x);
#else
    float e = __expf(2.0f * x);                     // v_exp_f32 (pre-scaled by log2e)
    float r = __builtin_amdgcn_rcpf(e + 1.0f);      // single v_rcp_f32
    return 1.0f - 2.0f * r;
#endif
}

// ---------- kernel 1: W_enc f32 -> bf16 (512*512 = 262144 elems) ----------
__global__ void k_convert_w(const float* __restrict__ w, __bf16* __restrict__ wb) {
    int i = blockIdx.x * 256 + threadIdx.x;
    wb[i] = (__bf16)w[i];
}

// ---------- kernel 2: row_bias[b][a] = dec[b]·W_dec[a] + W_dec_b + W_enc_b + bias ----------
__global__ void k_row_bias(const float* __restrict__ dec, const float* __restrict__ Wdec,
                           const float* __restrict__ Wdb, const float* __restrict__ Web,
                           const float* __restrict__ bias, float* __restrict__ rb) {
    __shared__ float sdec[ND];
    int b = blockIdx.x, tid = threadIdx.x;
    sdec[tid]       = dec[b * ND + tid];
    sdec[tid + 256] = dec[b * ND + tid + 256];
    __syncthreads();
    #pragma unroll
    for (int j = 0; j < 2; ++j) {
        int a = tid + j * 256;
        const float* wr = Wdec + (size_t)a * ND;
        float acc = Wdb[a] + Web[a] + bias[a];
        for (int d = 0; d < ND; ++d) acc += sdec[d] * wr[d];
        rb[b * ND + a] = acc;
    }
}

// ---------- kernel 3: scores[b][s] = V·tanh(enc·W_encT + row_bias) + V_b  (WMMA bf16) ----------
// grid: (S/64, B), block: 256 (8 waves). Each wave: 64 s-rows x 64 ATT cols (4x4 WMMA tiles).
__global__ void __launch_bounds__(256, 1)
k_scores(const float* __restrict__ enc, const __bf16* __restrict__ Wb,
         const float* __restrict__ rbias, const float* __restrict__ Vw,
         const float* __restrict__ Vb, float* __restrict__ scores) {
    __shared__ __align__(16) __bf16 ldsA[MT * ND];   // 64 rows x 512 K, bf16 (64 KB)
    __shared__ float red[8][MT];

    const int stile = blockIdx.x;
    const int b     = blockIdx.y;
    const int tid   = threadIdx.x;

    // stage 64x512 enc tile to LDS as bf16 (float4 reads, 8B bf16x4 stores)
    const float4* src4 = (const float4*)(enc + ((size_t)b * NS + (size_t)stile * MT) * ND);
    #pragma unroll
    for (int i = 0; i < (MT * ND / 4) / 256; ++i) {   // 32 iters
        int idx = tid + i * 256;                      // float4 index
        float4 f = src4[idx];
        bf16x4 o;
        o.x = (__bf16)f.x; o.y = (__bf16)f.y; o.z = (__bf16)f.z; o.w = (__bf16)f.w;
        *(bf16x4*)&ldsA[idx * 4] = o;
    }
    __syncthreads();

    const int wave  = tid >> 5;
    const int lane  = tid & 31;
    const int l16   = lane & 15;
    const bool hi   = lane >= 16;
    const int nbase = wave * 64;

    const v8f vzero = {0.f,0.f,0.f,0.f,0.f,0.f,0.f,0.f};
    v8f acc[4][4];
    #pragma unroll
    for (int mt = 0; mt < 4; ++mt)
        #pragma unroll
        for (int nt = 0; nt < 4; ++nt) acc[mt][nt] = vzero;

    for (int k0 = 0; k0 < ND; k0 += 32) {
        // A fragments (16x32 bf16): lanes 0-15 row=l16 K={k0..k0+7, k0+16..23};
        // lanes 16-31 same row, K={k0+8..15, k0+24..31}
        const int e0 = k0 + (hi ? 8 : 0);
        v16bf afrag[4];
        #pragma unroll
        for (int mt = 0; mt < 4; ++mt) {
            const __bf16* ap = &ldsA[(mt * 16 + l16) * ND + e0];
            afrag[mt] = make_frag(ap, ap + 16);
        }
        // B fragments (32x16 bf16): lanes 0-15 col=l16 K={k0..k0+7, k0+16..23}; lanes 16-31 K={k0+8..15, k0+24..31}
        const int kb = k0 + (hi ? 8 : 0);
        #pragma unroll
        for (int nt = 0; nt < 4; ++nt) {
            const __bf16* wp = Wb + (size_t)(nbase + nt * 16 + l16) * ND + kb;
            v16bf bfrag = make_frag(wp, wp + 16);
            #pragma unroll
            for (int mt = 0; mt < 4; ++mt)
                acc[mt][nt] = __builtin_amdgcn_wmma_f32_16x16x32_bf16(
                    false, afrag[mt], false, bfrag, (short)0, acc[mt][nt], false, false);
        }
    }

    // epilogue: per M-tile, reduce tanh(acc + row_bias)*V_w over this wave's 64 columns
    float rbv[4], vwv[4];
    #pragma unroll
    for (int nt = 0; nt < 4; ++nt) {
        const int aIdx = nbase + nt * 16 + l16;
        rbv[nt] = rbias[b * ND + aIdx];
        vwv[nt] = Vw[aIdx];
    }
    #pragma unroll
    for (int mt = 0; mt < 4; ++mt) {
        float p[8] = {0.f,0.f,0.f,0.f,0.f,0.f,0.f,0.f};
        #pragma unroll
        for (int nt = 0; nt < 4; ++nt) {
            #pragma unroll
            for (int r = 0; r < 8; ++r)
                p[r] += fast_tanh(acc[mt][nt][r] + rbv[nt]) * vwv[nt];
        }
        // reduce across the 16 lanes of each half (C layout: N = lane%16, M = r + 8*hi)
        #pragma unroll
        for (int r = 0; r < 8; ++r) {
            float v = p[r];
            v += __shfl_xor(v, 1, 32);
            v += __shfl_xor(v, 2, 32);
            v += __shfl_xor(v, 4, 32);
            v += __shfl_xor(v, 8, 32);
            p[r] = v;
        }
        if (l16 == 0) {
            const int mof = mt * 16 + (hi ? 8 : 0);
            #pragma unroll
            for (int r = 0; r < 8; ++r) red[wave][mof + r] = p[r];
        }
    }
    __syncthreads();
    // cross-wave reduce over the 8 ATT column groups
    if (tid < MT) {
        float t = Vb[0];
        #pragma unroll
        for (int w = 0; w < 8; ++w) t += red[w][tid];
        scores[(size_t)b * NS + (size_t)stile * MT + tid] = t;
    }
}

// ---------- kernel 4: softmax over S, in place ----------
__global__ void k_softmax(float* __restrict__ sc) {
    __shared__ float rmax[8];
    __shared__ float rsum[8];
    const int b = blockIdx.x, tid = threadIdx.x;
    const int wave = tid >> 5, lane = tid & 31;
    float* row = sc + (size_t)b * NS;

    float v[16];
    float m = -1e30f;
    #pragma unroll
    for (int i = 0; i < 16; ++i) { v[i] = row[tid + i * 256]; m = fmaxf(m, v[i]); }
    #pragma unroll
    for (int off = 1; off < 32; off <<= 1) m = fmaxf(m, __shfl_xor(m, off, 32));
    if (lane == 0) rmax[wave] = m;
    __syncthreads();
    m = rmax[0];
    #pragma unroll
    for (int w = 1; w < 8; ++w) m = fmaxf(m, rmax[w]);

    float s = 0.f;
    #pragma unroll
    for (int i = 0; i < 16; ++i) { v[i] = __expf(v[i] - m); s += v[i]; }
    #pragma unroll
    for (int off = 1; off < 32; off <<= 1) s += __shfl_xor(s, off, 32);
    if (lane == 0) rsum[wave] = s;
    __syncthreads();
    s = 0.f;
    #pragma unroll
    for (int w = 0; w < 8; ++w) s += rsum[w];
    const float inv = 1.0f / s;
    #pragma unroll
    for (int i = 0; i < 16; ++i) row[tid + i * 256] = v[i] * inv;
}

// ---------- kernel 5: context partials over s-chunks of 256 ----------
__global__ void k_ctx_partial(const float* __restrict__ enc, const float* __restrict__ attn,
                              float* __restrict__ part) {
    __shared__ float sat[256];
    const int b = blockIdx.x, chunk = blockIdx.y, tid = threadIdx.x;
    const float* encb = enc + ((size_t)b * NS + (size_t)chunk * 256) * ND;
    sat[tid] = attn[(size_t)b * NS + (size_t)chunk * 256 + tid];
    __syncthreads();
    float a0 = 0.f, a1 = 0.f;
    for (int s = 0; s < 256; ++s) {
        const float w = sat[s];
        a0 += w * encb[(size_t)s * ND + tid];
        a1 += w * encb[(size_t)s * ND + tid + 256];
    }
    float* pp = part + ((size_t)b * 16 + chunk) * ND;
    pp[tid]       = a0;
    pp[tid + 256] = a1;
}

// ---------- kernel 6: reduce partials -> context ----------
__global__ void k_ctx_reduce(const float* __restrict__ part, float* __restrict__ ctx) {
    const int b = blockIdx.x, tid = threadIdx.x;
    float a0 = 0.f, a1 = 0.f;
    #pragma unroll
    for (int c = 0; c < 16; ++c) {
        const float* pp = part + ((size_t)b * 16 + c) * ND;
        a0 += pp[tid];
        a1 += pp[tid + 256];
    }
    ctx[(size_t)b * ND + tid]       = a0;
    ctx[(size_t)b * ND + tid + 256] = a1;
}

extern "C" void kernel_launch(void* const* d_in, const int* in_sizes, int n_in,
                              void* d_out, int out_size, void* d_ws, size_t ws_size,
                              hipStream_t stream) {
    const float* enc  = (const float*)d_in[0];  // [32,4096,512]
    const float* dec  = (const float*)d_in[1];  // [32,512]
    const float* Wenc = (const float*)d_in[2];  // [512,512]
    const float* Web  = (const float*)d_in[3];  // [512]
    const float* Wdec = (const float*)d_in[4];  // [512,512]
    const float* Wdb  = (const float*)d_in[5];  // [512]
    const float* Vw   = (const float*)d_in[6];  // [1,512]
    const float* Vb   = (const float*)d_in[7];  // [1]
    const float* bias = (const float*)d_in[8];  // [512]

    float* out  = (float*)d_out;
    float* ctx  = out;               // [32,512]   = 16384 floats
    float* attn = out + NB * ND;     // [32,4096]  = 131072 floats

    char* ws = (char*)d_ws;
    __bf16* Wb   = (__bf16*)ws;                       // 512 KB
    float*  rb   = (float*)(ws + 512 * 1024);         //  64 KB
    float*  part = (float*)(ws + 576 * 1024);         //   1 MB (32*16*512 f32)

    k_convert_w  <<<1024, 256, 0, stream>>>(Wenc, Wb);
    k_row_bias   <<<NB,   256, 0, stream>>>(dec, Wdec, Wdb, Web, bias, rb);
    k_scores     <<<dim3(NS / MT, NB), 256, 0, stream>>>(enc, Wb, rb, Vw, Vb, attn);
    k_softmax    <<<NB,   256, 0, stream>>>(attn);
    k_ctx_partial<<<dim3(NB, 16), 256, 0, stream>>>(enc, attn, part);
    k_ctx_reduce <<<NB,   256, 0, stream>>>(part, ctx);
}
